// SelfAttentionBlock_20787641712972
// MI455X (gfx1250) — matrix operand order repaired and checked by reading, not compiled, not run on hardware
//
#include <hip/hip_runtime.h>
#include <hip/hip_bf16.h>

typedef __bf16 bf16;
typedef __attribute__((ext_vector_type(16))) __bf16 v16bf;
typedef __attribute__((ext_vector_type(8)))  __bf16 v8bf;
typedef __attribute__((ext_vector_type(8)))  float   v8f;
typedef __attribute__((ext_vector_type(4)))  float   vf4;
typedef __attribute__((ext_vector_type(4)))  unsigned int u32x4;
typedef __attribute__((ext_vector_type(8)))  int          i32x8;
typedef __attribute__((ext_vector_type(4)))  int          i32x4;

#define N_   8
#define C_   256
#define HW_  4096
#define CHW_ (C_*HW_)
#define EPS_ 1e-5f

#define WMMA_BF16(a,b,c) __builtin_amdgcn_wmma_f32_16x16x32_bf16(false,(a),false,(b),(short)0,(c),false,false)

// ---- explicit global-address-space accessors (avoid flat_load: flat couples
// LOADcnt with DScnt and serializes the LDS->WMMA pipeline) ------------------
template <typename T>
__device__ __forceinline__ T gload(const void* p) {
  return *(const __attribute__((address_space(1))) T*)(__UINTPTR_TYPE__)p;
}
template <typename T>
__device__ __forceinline__ void gstore(void* p, T v) {
  *(__attribute__((address_space(1))) T*)(__UINTPTR_TYPE__)p = v;
}

// ---- fragment loaders (per CDNA5 ISA 7.12.2 layouts) -----------------------
// A 16x32 bf16, tile row-major [M][K], pitch multiple of 8 elements.
__device__ __forceinline__ v16bf frag_a(const bf16* tile, int pitch, int row0, int k0) {
  const int lane = threadIdx.x & 31;
  const int m  = lane & 15;
  const int kb = (lane >> 4) << 3;          // lanes 16-31 start at K=+8
  const bf16* p = tile + (row0 + m) * pitch + k0 + kb;
  v8bf lo = *(const v8bf*)(p);              // K = kb .. kb+7
  v8bf hi = *(const v8bf*)(p + 16);         // K = kb+16 .. kb+23
  v16bf r;
#pragma unroll
  for (int t = 0; t < 8; ++t) { r[t] = lo[t]; r[t + 8] = hi[t]; }
  return r;
}

// B 32x16 bf16, tile stored TRANSPOSED row-major [N][K].
__device__ __forceinline__ v16bf frag_b(const bf16* tileT, int pitch, int n0, int k0) {
  const int lane = threadIdx.x & 31;
  const int n  = lane & 15;
  const int kb = (lane >> 4) << 3;          // lanes 16-31 start at K=+8
  const bf16* p = tileT + (n0 + n) * pitch + k0 + kb;
  v8bf lo = *(const v8bf*)(p);
  v8bf hi = *(const v8bf*)(p + 16);
  v16bf r;
#pragma unroll
  for (int t = 0; t < 8; ++t) { r[t] = lo[t]; r[t + 8] = hi[t]; }
  return r;
}

// ---- LayerNorm stats (deterministic two-pass) ------------------------------
__global__ void ln_stats_partial(const float* __restrict__ x, float2* __restrict__ part) {
  const int n = blockIdx.y, b = blockIdx.x, tid = threadIdx.x;
  const float* px = x + (size_t)n * CHW_ + b * 8192;
  float s = 0.f, sq = 0.f;
#pragma unroll
  for (int it = 0; it < 8; ++it) {
    vf4 v = gload<vf4>(px + it * 1024 + tid * 4);
    s  += v.x + v.y + v.z + v.w;
    sq += v.x * v.x + v.y * v.y + v.z * v.z + v.w * v.w;
  }
  __shared__ float rs[256], rq[256];
  rs[tid] = s; rq[tid] = sq; __syncthreads();
  for (int off = 128; off > 0; off >>= 1) {
    if (tid < off) { rs[tid] += rs[tid + off]; rq[tid] += rq[tid + off]; }
    __syncthreads();
  }
  if (tid == 0) part[n * 128 + b] = make_float2(rs[0], rq[0]);
}

__global__ void ln_stats_final(const float2* __restrict__ part, float2* __restrict__ stats) {
  const int n = blockIdx.x, tid = threadIdx.x;
  __shared__ float rs[128], rq[128];
  float2 v = part[n * 128 + tid];
  rs[tid] = v.x; rq[tid] = v.y; __syncthreads();
  for (int off = 64; off > 0; off >>= 1) {
    if (tid < off) { rs[tid] += rs[tid + off]; rq[tid] += rq[tid + off]; }
    __syncthreads();
  }
  if (tid == 0) {
    float mean = rs[0] * (1.0f / (float)CHW_);
    float var  = rq[0] * (1.0f / (float)CHW_) - mean * mean;
    stats[n] = make_float2(mean, rsqrtf(var + EPS_));
  }
}

// ---- QKV projection: LN-normalize tile + 3 GEMMs vs 256x256 weights --------
__global__ __launch_bounds__(256)
void qkv_proj(const float* __restrict__ x, const float* __restrict__ lnw, const float* __restrict__ lnb,
              const float* __restrict__ wq, const float* __restrict__ bq,
              const float* __restrict__ wk, const float* __restrict__ bk,
              const float* __restrict__ wv, const float* __restrict__ bv,
              const float2* __restrict__ stats,
              bf16* __restrict__ Qg, bf16* __restrict__ Kg, bf16* __restrict__ Vg) {
  const int n = blockIdx.y;
  const int ibase = blockIdx.x * 64;
  const int tid = threadIdx.x;
  __shared__ __align__(16) bf16 sNX[64 * 264];   // nx^T tile [i][c]
  __shared__ __align__(16) bf16 sW[64 * 264];    // weight chunk [cout][c]

  const float2 st = stats[n];
  const float mean = st.x, rstd = st.y;

  for (int it = 0; it < 64; ++it) {              // normalize + transpose into LDS
    int idx = it * 256 + tid;
    int c = idx >> 6, i = idx & 63;
    size_t g = (size_t)c * HW_ + ibase + i;
    float nv = (gload<float>(x + (size_t)n * CHW_ + g) - mean) * rstd
               * gload<float>(lnw + g) + gload<float>(lnb + g);
    sNX[i * 264 + c] = (bf16)nv;
  }
  __syncthreads();

  const float* Ws[3] = {wq, wk, wv};
  const float* Bs[3] = {bq, bk, bv};
  bf16*        Os[3] = {Qg, Kg, Vg};

  const int w = tid >> 5;
  const int lane = tid & 31;
  const int h2 = lane >> 4;
  const int ln15 = lane & 15;

#pragma unroll                                   // unroll => direct kernarg ptrs
  for (int p = 0; p < 3; ++p) {
    const float* W = Ws[p];
    const float* B = Bs[p];
    bf16* O = Os[p];
    for (int cc = 0; cc < 4; ++cc) {
      for (int it = 0; it < 16; ++it) {          // 64x256 f32 chunk -> bf16 LDS
        int e = (it * 256 + tid) * 4;
        int co = e >> 8, c0 = e & 255;
        vf4 v = gload<vf4>(W + (size_t)(cc * 64 + co) * C_ + c0);
        sW[co * 264 + c0 + 0] = (bf16)v.x;
        sW[co * 264 + c0 + 1] = (bf16)v.y;
        sW[co * 264 + c0 + 2] = (bf16)v.z;
        sW[co * 264 + c0 + 3] = (bf16)v.w;
      }
      __syncthreads();
#pragma unroll
      for (int q = 0; q < 2; ++q) {
        int f = w * 2 + q;
        int fm = f >> 2, fn = f & 3;             // fm: cout frag, fn: i frag
        v8f acc = {0.f,0.f,0.f,0.f,0.f,0.f,0.f,0.f};
#pragma unroll
        for (int kk = 0; kk < 8; ++kk) {
          v16bf a = frag_a(sW,  264, fm * 16, kk * 32);
          v16bf b = frag_b(sNX, 264, fn * 16, kk * 32);
          acc = WMMA_BF16(a, b, acc);
        }
        int iglob  = ibase + fn * 16 + ln15;     // D col
        int cobase = cc * 64 + fm * 16 + h2 * 8; // D rows r..r+7 contiguous
        v8bf ov;
#pragma unroll
        for (int r = 0; r < 8; ++r) ov[r] = (bf16)(acc[r] + gload<float>(B + cobase + r));
        gstore<v8bf>(O + ((size_t)n * HW_ + iglob) * C_ + cobase, ov);
      }
      __syncthreads();
    }
  }
}

// ---- TDM: load one [64][256] bf16 tile (contiguous 32KB) into LDS with an
// 8-element row pad so the LDS pitch stays 264 (bank-conflict-free) ----------
#if __has_builtin(__builtin_amdgcn_tensor_load_to_lds)
#define HAS_TDM 1
__device__ __forceinline__ void tdm_load_k_tile(const bf16* gsrc, bf16* ldst) {
  unsigned ldsb = (unsigned)(__UINTPTR_TYPE__)(void*)ldst;   // low 32b = LDS offset
  unsigned long long ga = (unsigned long long)(__UINTPTR_TYPE__)(const void*)gsrc;
  u32x4 g0;
  g0[0] = 1u;                                   // count=1, user descriptor
  g0[1] = ldsb;                                 // lds_addr
  g0[2] = (unsigned)ga;                         // global_addr[31:0]
  g0[3] = (unsigned)((ga >> 32) & 0x01FFFFFFu) | (2u << 30);  // ga[56:32] | type=2
  i32x8 g1;
  g1[0] = (int)((1u << 16) |                    // data_size = 2B
                (1u << 20) |                    // pad_enable
                (6u << 22) |                    // pad_interval: 128 dwords (512B row)
                (3u << 25));                    // pad_amount: 4 dwords (16B pad)
  g1[1] = (int)(256u << 16);                    // tensor_dim0 = 256
  g1[2] = (int)(4096u << 16);                   // tensor_dim1 = 4096
  g1[3] = (int)(256u << 16);                    // tile_dim0 = 256
  g1[4] = 64;                                   // tile_dim1 = 64
  g1[5] = 256;                                  // tensor_dim0_stride = 256
  g1[6] = 0; g1[7] = 0;
  i32x4 z4 = {0, 0, 0, 0};
  i32x8 z8 = {0, 0, 0, 0, 0, 0, 0, 0};
  // 6-arg form (clang-23 / therock headers): (g0, g1, g2, g3, g4, cpol)
  __builtin_amdgcn_tensor_load_to_lds(g0, g1, z4, z4, z8, 0);
}
#else
#define HAS_TDM 0
#endif

// ---- fused flash attention + output projection + residual ------------------
__global__ __launch_bounds__(256)
void attn_out(const float* __restrict__ x, const float* __restrict__ wo, const float* __restrict__ bo,
              const bf16* __restrict__ Qg, const bf16* __restrict__ Kg, const bf16* __restrict__ Vg,
              float* __restrict__ out) {
  const int n = blockIdx.y;
  const int ibase = blockIdx.x * 64;
  const int tid = threadIdx.x;
  const int w = tid >> 5;
  const int lane = tid & 31;
  const int h2 = lane >> 4;
  const int ln15 = lane & 15;

  __shared__ __align__(16) char smem[166656];
  bf16*  sQ  = (bf16*)(smem);                // [64][264] Q tile [i][c]
  bf16*  sK0 = (bf16*)(smem + 33792);        // [64][264] K tile buf0 (later O tile)
  bf16*  sK1 = (bf16*)(smem + 67584);        // [64][264] K tile buf1
  bf16*  sV  = (bf16*)(smem + 101376);       // [256][72] V^T tile (later wo chunk)
  float* sS  = (float*)(smem + 138240);      // [64][72] scores
  bf16*  sP  = (bf16*)(smem + 156672);       // [64][72] probabilities
  float* sM  = (float*)(smem + 165888);
  float* sL  = (float*)(smem + 166144);
  float* sA  = (float*)(smem + 166400);
  bf16* sKb[2] = {sK0, sK1};

  const bf16* Kbase = Kg + (size_t)n * HW_ * C_;

#if HAS_TDM
  if (tid < 32) tdm_load_k_tile(Kbase, sK0);   // prefetch K tile jt=0
#endif

#pragma unroll
  for (int it = 0; it < 8; ++it) {             // load Q tile
    int e = (it * 256 + tid) * 8;
    int i = e >> 8, c = e & 255;
    *(v8bf*)(sQ + i * 264 + c) =
        gload<v8bf>(Qg + ((size_t)n * HW_ + ibase + i) * C_ + c);
  }
  if (tid < 64) { sM[tid] = -3.0e38f; sL[tid] = 0.f; }

  const int fi  = w & 3;                       // i-fragment row for O accumulators
  const int fcb = (w >> 2) * 8;                // c-fragment base (0 or 8)
  v8f oacc[8];
#pragma unroll
  for (int f = 0; f < 8; ++f)
#pragma unroll
    for (int r = 0; r < 8; ++r) oacc[f][r] = 0.f;

  for (int jt = 0; jt < 64; ++jt) {
    const int jbase = jt * 64;
    bf16* sK = sKb[jt & 1];

#pragma unroll
    for (int it = 0; it < 8; ++it) {           // V^T tile [c][j]
      int e = (it * 256 + tid) * 8;
      int j = e >> 8, c0 = e & 255;
      v8bf v = gload<v8bf>(Vg + ((size_t)n * HW_ + jbase + j) * C_ + c0);
#pragma unroll
      for (int t = 0; t < 8; ++t) sV[(c0 + t) * 72 + j] = v[t];
    }

#if HAS_TDM
    if (tid < 32) {                            // wave 0 drives the TDM pipeline
      if (jt < 63) {
        tdm_load_k_tile(Kbase + (size_t)(jbase + 64) * C_, sKb[(jt + 1) & 1]);
        __builtin_amdgcn_s_wait_tensorcnt(1);  // retire tile jt (in-order)
      } else {
        __builtin_amdgcn_s_wait_tensorcnt(0);
      }
    }
#else
#pragma unroll
    for (int it = 0; it < 8; ++it) {           // cooperative K tile copy
      int e = (it * 256 + tid) * 8;
      int j = e >> 8, c = e & 255;
      *(v8bf*)(sK + j * 264 + c) = gload<v8bf>(Kbase + (size_t)(jbase + j) * C_ + c);
    }
#endif
    __syncthreads();

    // S = (Q K^T) * 1/sqrt(C)
#pragma unroll
    for (int q = 0; q < 2; ++q) {
      int f = w * 2 + q;
      int fm = f >> 2, fn = f & 3;
      v8f acc = {0.f,0.f,0.f,0.f,0.f,0.f,0.f,0.f};
#pragma unroll
      for (int kk = 0; kk < 8; ++kk) {
        v16bf a = frag_a(sQ, 264, fm * 16, kk * 32);
        v16bf b = frag_b(sK, 264, fn * 16, kk * 32);
        acc = WMMA_BF16(a, b, acc);
      }
      int col   = fn * 16 + ln15;
      int rbase = fm * 16 + h2 * 8;
#pragma unroll
      for (int r = 0; r < 8; ++r) sS[(rbase + r) * 72 + col] = acc[r] * 0.0625f;
    }
    __syncthreads();

    if (tid < 64) {                            // online softmax, 1 thread / row
      const int row = tid;
      float mo = sM[row];
      float tmax = -3.0e38f;
#pragma unroll 8
      for (int j = 0; j < 64; ++j) tmax = fmaxf(tmax, sS[row * 72 + j]);
      float mn = fmaxf(mo, tmax);
      float alpha = __expf(mo - mn);
      float tsum = 0.f;
#pragma unroll 8
      for (int j = 0; j < 64; ++j) {
        float pv = __expf(sS[row * 72 + j] - mn);
        sP[row * 72 + j] = (bf16)pv;
        tsum += pv;
      }
      sM[row] = mn;
      sL[row] = sL[row] * alpha + tsum;
      sA[row] = alpha;
    }
    __syncthreads();

    // O = alpha*O + P V^T
    float ar[8];
#pragma unroll
    for (int r = 0; r < 8; ++r) ar[r] = sA[fi * 16 + h2 * 8 + r];
#pragma unroll
    for (int f = 0; f < 8; ++f)
#pragma unroll
      for (int r = 0; r < 8; ++r) oacc[f][r] *= ar[r];
    v16bf a0 = frag_a(sP, 72, fi * 16, 0);
    v16bf a1 = frag_a(sP, 72, fi * 16, 32);
#pragma unroll
    for (int f = 0; f < 8; ++f) {
      v16bf b0 = frag_b(sV, 72, (fcb + f) * 16, 0);
      oacc[f] = WMMA_BF16(a0, b0, oacc[f]);
      v16bf b1 = frag_b(sV, 72, (fcb + f) * 16, 32);
      oacc[f] = WMMA_BF16(a1, b1, oacc[f]);
    }
    __syncthreads();
  }

  // finalize O /= l, stage as bf16 tile [i][c] (reuse K buf0)
  bf16* sO = sK0;
  float lr[8];
#pragma unroll
  for (int r = 0; r < 8; ++r) lr[r] = 1.0f / sL[fi * 16 + h2 * 8 + r];
#pragma unroll
  for (int f = 0; f < 8; ++f) {
    int c = (fcb + f) * 16 + ln15;
#pragma unroll
    for (int r = 0; r < 8; ++r)
      sO[(fi * 16 + h2 * 8 + r) * 264 + c] = (bf16)(oacc[f][r] * lr[r]);
  }
  __syncthreads();

  // output projection + bias + residual
  bf16* sWO = sV;                              // reuse as [64][264] wo chunk
  for (int cc = 0; cc < 4; ++cc) {
    for (int it = 0; it < 16; ++it) {
      int e = (it * 256 + tid) * 4;
      int co = e >> 8, c0 = e & 255;
      vf4 v = gload<vf4>(wo + (size_t)(cc * 64 + co) * C_ + c0);
      sWO[co * 264 + c0 + 0] = (bf16)v.x;
      sWO[co * 264 + c0 + 1] = (bf16)v.y;
      sWO[co * 264 + c0 + 2] = (bf16)v.z;
      sWO[co * 264 + c0 + 3] = (bf16)v.w;
    }
    __syncthreads();
#pragma unroll
    for (int q = 0; q < 2; ++q) {
      int f = w * 2 + q;
      int fm = f >> 2, fn = f & 3;             // fm: i frag, fn: cout frag
      v8f acc = {0.f,0.f,0.f,0.f,0.f,0.f,0.f,0.f};
#pragma unroll
      for (int kk = 0; kk < 8; ++kk) {
        v16bf a = frag_a(sO,  264, fm * 16, kk * 32);
        v16bf b = frag_b(sWO, 264, fn * 16, kk * 32);
        acc = WMMA_BF16(a, b, acc);
      }
      int co = cc * 64 + fn * 16 + ln15;
      int i0 = ibase + fm * 16 + h2 * 8;
      const float* xr = x + ((size_t)n * C_ + co) * HW_ + i0;
      float* orow = out + ((size_t)n * C_ + co) * HW_ + i0;
      float bov = gload<float>(bo + co);
      vf4 x0 = gload<vf4>(xr);
      vf4 x1 = gload<vf4>(xr + 4);
      vf4 o0, o1;
      o0.x = x0.x + acc[0] + bov; o0.y = x0.y + acc[1] + bov;
      o0.z = x0.z + acc[2] + bov; o0.w = x0.w + acc[3] + bov;
      o1.x = x1.x + acc[4] + bov; o1.y = x1.y + acc[5] + bov;
      o1.z = x1.z + acc[6] + bov; o1.w = x1.w + acc[7] + bov;
      gstore<vf4>(orow, o0);
      gstore<vf4>(orow + 4, o1);
    }
    __syncthreads();
  }
}

extern "C" void kernel_launch(void* const* d_in, const int* in_sizes, int n_in,
                              void* d_out, int out_size, void* d_ws, size_t ws_size,
                              hipStream_t stream) {
  const float* x   = (const float*)d_in[0];
  const float* lnw = (const float*)d_in[1];
  const float* lnb = (const float*)d_in[2];
  const float* wq  = (const float*)d_in[3];
  const float* bq  = (const float*)d_in[4];
  const float* wk  = (const float*)d_in[5];
  const float* bk  = (const float*)d_in[6];
  const float* wv  = (const float*)d_in[7];
  const float* bv  = (const float*)d_in[8];
  const float* wo  = (const float*)d_in[9];
  const float* bo  = (const float*)d_in[10];
  float* out = (float*)d_out;

  char* ws = (char*)d_ws;
  float2* p_part  = (float2*)ws;                     // 8*128 float2
  float2* p_stats = (float2*)(ws + 8192);            // 8 float2
  bf16* Qg = (bf16*)(ws + 16384);                    // [8][4096][256] bf16
  bf16* Kg = Qg + (size_t)N_ * HW_ * C_;
  bf16* Vg = Kg + (size_t)N_ * HW_ * C_;

  ln_stats_partial<<<dim3(128, N_), 256, 0, stream>>>(x, p_part);
  ln_stats_final<<<N_, 128, 0, stream>>>(p_part, p_stats);
  qkv_proj<<<dim3(64, N_), 256, 0, stream>>>(x, lnw, lnb, wq, bq, wk, bk, wv, bv,
                                             p_stats, Qg, Kg, Vg);
  attn_out<<<dim3(64, N_), 256, 0, stream>>>(x, wo, bo, Qg, Kg, Vg, out);
}
